// InvSlotAttentionGuide_87978110091991
// MI455X (gfx1250) — compile-verified, matrix operand order checked by reading, not run on hardware
//
#include <hip/hip_runtime.h>
#include <hip/hip_bf16.h>

// ---------------------------------------------------------------------------
// Types / helpers
// ---------------------------------------------------------------------------
typedef __attribute__((ext_vector_type(16))) _Float16 v16h;
typedef __attribute__((ext_vector_type(8)))  _Float16 v8h;
typedef __attribute__((ext_vector_type(8)))  float    v8f;

union Frag16 { v16h v16; v8h v8[2]; };

// WMMA 16-bit fragment (16x32, row x K) loaded as two contiguous 16B runs.
// Per CDNA5 ISA 7.12.2: lanes 0-15 hold row (lane&15), K 0..7 & 16..23;
// lanes 16-31 hold K 8..15 & 24..31.  `p` must point at (row, k0) already.
__device__ inline v16h frag_ld(const _Float16* p) {
    Frag16 f;
    f.v8[0] = *(const v8h*)(p);
    f.v8[1] = *(const v8h*)(p + 16);
    return f.v16;
}

// Same fragment gather but from an LDS tile stored row-major [16][32].
__device__ inline v16h frag_ld_lds(const _Float16* tile, int lane) {
    const int r  = lane & 15;
    const int k0 = (lane & 16) ? 8 : 0;
    Frag16 f;
    f.v8[0] = *(const v8h*)(tile + r * 32 + k0);
    f.v8[1] = *(const v8h*)(tile + r * 32 + k0 + 16);
    return f.v16;
}

__device__ inline float warp_sum_all(float v) {
    for (int o = 16; o > 0; o >>= 1) v += __shfl_xor(v, o);
    return v;
}

// Wave-private LDS tiles: same-wave DS ops complete in order on CDNA5, so a
// compiler scheduling fence is all that is needed (no block barrier).
__device__ inline void wave_lds_fence() {
    asm volatile("" ::: "memory");
    __builtin_amdgcn_wave_barrier();
}

#define WMMA_F16(A, B, C) \
    __builtin_amdgcn_wmma_f32_16x16x32_f16(false, (A), false, (B), (short)0, (C), false, false)

// ---------------------------------------------------------------------------
// Per-layer k -> (coff, dy, dx, valid) table for implicit-GEMM conv gather.
// ---------------------------------------------------------------------------
__global__ void build_ktab_k(int H, int W, int K, int Kp, int4* __restrict__ tab) {
    int k = blockIdx.x * blockDim.x + threadIdx.x;
    if (k >= Kp) return;
    if (k < K) {
        int ci = k / 25;
        int rr = k - ci * 25;
        int dy = rr / 5 - 2;
        int dx = rr - (rr / 5) * 5 - 2;
        tab[k] = make_int4(ci * H * W + dy * W + dx, dy, dx, 1);
    } else {
        tab[k] = make_int4(0, 0, 0, 0);
    }
}

// ---------------------------------------------------------------------------
// Implicit-GEMM 5x5 conv (pad=2), Cout==64, WMMA f16->f32, fused bias+ReLU.
//   in  : f16 [B][Cin][H][W]
//   wt  : f16 [64][Kp]  (zero-padded to Kp = ceil(Cin*25/32)*32)
//   out : f16 [B][64][oH][oW]
// One wave computes a 64(M) x 16(N) strip: one gathered B tile -> 4 WMMAs.
// Position info lives in registers and is broadcast with wave32 shuffles.
// ---------------------------------------------------------------------------
__global__ __launch_bounds__(256) void conv5x5_wmma(
    const _Float16* __restrict__ in, const _Float16* __restrict__ wt,
    const float* __restrict__ bias, const int4* __restrict__ ktab,
    _Float16* __restrict__ out,
    int Bn, int Cin, int H, int W, int stride, int oH, int oW, int Kp)
{
    __shared__ __align__(16) _Float16 ldsB[8][16 * 32];
    const int lane = threadIdx.x & 31;
    const int wid  = threadIdx.x >> 5;
    const int HW   = oH * oW;
    const int Ntot = Bn * HW;
    const int tile = blockIdx.x * 8 + wid;   // one 16-position tile per wave
    const int n0   = tile * 16;

    // Each lane decodes the position of column (lane & 15); kept in VGPRs.
    const int pn   = n0 + (lane & 15);
    const int pval = (pn < Ntot) ? 1 : 0;
    const int pc   = pval ? pn : 0;
    const int pb   = pc / HW;
    const int prem = pc - pb * HW;
    const int poy  = prem / oW;
    const int pox  = prem - poy * oW;
    const int piy  = poy * stride;
    const int pix  = pox * stride;
    const int pbase = ((pb * Cin) * H + piy) * W + pix;

    v8f acc0 = {}, acc1 = {}, acc2 = {}, acc3 = {};
    const int r  = lane & 15;
    const int k0 = (lane & 16) ? 8 : 0;
    const _Float16* wp = wt + r * Kp + k0;   // row r of weight tile j at +16*j*Kp

    for (int kb = 0; kb < Kp; kb += 32) {
        // lane <-> k assignment; all k geometry comes from the table
        const int4 kt = ktab[kb + lane];     // {coff, dy, dx, valid}
        for (int nl = 0; nl < 16; ++nl) {
            const int base = __shfl(pbase, nl);
            const int iy   = __shfl(piy, nl) + kt.y;
            const int ix   = __shfl(pix, nl) + kt.z;
            const int pv   = __shfl(pval, nl);
            _Float16 vb = (_Float16)0.f;
            if (kt.w && pv &&
                (unsigned)iy < (unsigned)H && (unsigned)ix < (unsigned)W)
                vb = in[base + kt.x];
            ldsB[wid][nl * 32 + lane] = vb;
        }
        wave_lds_fence();
        v16h fb  = frag_ld_lds(ldsB[wid], lane);
        v16h fa0 = frag_ld(wp + kb);
        v16h fa1 = frag_ld(wp + 16 * Kp + kb);
        v16h fa2 = frag_ld(wp + 32 * Kp + kb);
        v16h fa3 = frag_ld(wp + 48 * Kp + kb);
        acc0 = WMMA_F16(fa0, fb, acc0);
        acc1 = WMMA_F16(fa1, fb, acc1);
        acc2 = WMMA_F16(fa2, fb, acc2);
        acc3 = WMMA_F16(fa3, fb, acc3);
        wave_lds_fence();
    }

    const int nn = n0 + (lane & 15);
    if (nn >= Ntot) return;
    const int b   = nn / HW;
    const int rem = nn - b * HW;
    const int mb  = (lane & 16) ? 8 : 0;
    _Float16* op = out + (b * 64) * HW + rem;
    for (int rr = 0; rr < 8; ++rr) {
        int m0 = mb + rr;
        float v0 = acc0[rr] + bias[m0];
        float v1 = acc1[rr] + bias[m0 + 16];
        float v2 = acc2[rr] + bias[m0 + 32];
        float v3 = acc3[rr] + bias[m0 + 48];
        op[(m0)      * HW] = (_Float16)(v0 > 0.f ? v0 : 0.f);
        op[(m0 + 16) * HW] = (_Float16)(v1 > 0.f ? v1 : 0.f);
        op[(m0 + 32) * HW] = (_Float16)(v2 > 0.f ? v2 : 0.f);
        op[(m0 + 48) * HW] = (_Float16)(v3 > 0.f ? v3 : 0.f);
    }
}

// ---------------------------------------------------------------------------
// Dense GEMM, zero staging: out[n,m] = act( sum_k X[n,k]*Wt[m,k] + bias[m] )
//   X f16 [Nrows][K], Wt f16 [M][K]; K % 32 == 0, M % 16 == 0, Nrows % 64 == 0.
// One wave computes 16(M) x 64(N): A + 4 B fragments direct from global.
// ---------------------------------------------------------------------------
__global__ __launch_bounds__(256) void gemm_wmma(
    const _Float16* __restrict__ X, const _Float16* __restrict__ Wt,
    const float* __restrict__ bias, float* __restrict__ outF,
    _Float16* __restrict__ outH, int Nrows, int K, int M, int relu)
{
    const int lane = threadIdx.x & 31;
    const int wid  = threadIdx.x >> 5;
    const int Mtiles = M >> 4;
    const int Nt64   = Nrows >> 6;
    const int tile = blockIdx.x * 8 + wid;
    if (tile >= Mtiles * Nt64) return;      // wave-uniform exit
    const int mt = tile % Mtiles;
    const int nt = tile / Mtiles;

    const int r  = lane & 15;
    const int k0 = (lane & 16) ? 8 : 0;
    const _Float16* wp = Wt + (mt * 16 + r) * K + k0;
    const _Float16* xp = X + ((long long)(nt * 64) + r) * K + k0;
    const long long s16 = (long long)16 * K;

    v8f acc0 = {}, acc1 = {}, acc2 = {}, acc3 = {};
    for (int kb = 0; kb < K; kb += 32) {
        v16h fa  = frag_ld(wp + kb);
        v16h fb0 = frag_ld(xp + kb);
        v16h fb1 = frag_ld(xp + s16 + kb);
        v16h fb2 = frag_ld(xp + 2 * s16 + kb);
        v16h fb3 = frag_ld(xp + 3 * s16 + kb);
        acc0 = WMMA_F16(fa, fb0, acc0);
        acc1 = WMMA_F16(fa, fb1, acc1);
        acc2 = WMMA_F16(fa, fb2, acc2);
        acc3 = WMMA_F16(fa, fb3, acc3);
    }

    const int mb = (lane & 16) ? 8 : 0;
    const long long nn0 = (long long)nt * 64 + (lane & 15);
    for (int rr = 0; rr < 8; ++rr) {
        const int m = mt * 16 + mb + rr;
        float v0 = acc0[rr], v1 = acc1[rr], v2 = acc2[rr], v3 = acc3[rr];
        if (bias) { float bv = bias[m]; v0 += bv; v1 += bv; v2 += bv; v3 += bv; }
        if (relu) {
            v0 = v0 > 0.f ? v0 : 0.f; v1 = v1 > 0.f ? v1 : 0.f;
            v2 = v2 > 0.f ? v2 : 0.f; v3 = v3 > 0.f ? v3 : 0.f;
        }
        if (outF) {
            outF[(nn0)      * M + m] = v0;
            outF[(nn0 + 16) * M + m] = v1;
            outF[(nn0 + 32) * M + m] = v2;
            outF[(nn0 + 48) * M + m] = v3;
        }
        if (outH) {
            outH[(nn0)      * M + m] = (_Float16)v0;
            outH[(nn0 + 16) * M + m] = (_Float16)v1;
            outH[(nn0 + 32) * M + m] = (_Float16)v2;
            outH[(nn0 + 48) * M + m] = (_Float16)v3;
        }
    }
}

// ---------------------------------------------------------------------------
// Fused MESH + Sinkhorn + attention update. One block per batch element.
// LDS-resident cost matrix C[1024][8] for the entire procedure:
//   C = 1 - kn.qn ; 5x( 5 Sinkhorn iters ); entropy-gradient step on C after
//   the first 4; final pi -> attn[b,s,n] and upd[b,s,d] = sum_n pi * v[n,d].
// ---------------------------------------------------------------------------
__global__ __launch_bounds__(256) void mesh_sinkhorn_k(
    const float* __restrict__ kn, const float* __restrict__ qn,
    const float* __restrict__ a, const float* __restrict__ bm,
    const float* __restrict__ vf, float* __restrict__ attn,
    float* __restrict__ upd)
{
    __shared__ float Cs[1024 * 8];
    __shared__ float us[1024];
    __shared__ float loga[1024];
    __shared__ float qs[8 * 64];
    __shared__ float wred[8][8];
    __shared__ float msh[8];
    __shared__ float vs[8];
    __shared__ float logb[8];

    const int b = blockIdx.x, tid = threadIdx.x;
    const int lane = tid & 31, wid = tid >> 5;

    for (int i = tid; i < 512; i += 256) qs[i] = qn[b * 512 + i];
    if (tid < 8) { logb[tid] = logf(bm[b * 8 + tid]); vs[tid] = 0.f; }
    __syncthreads();

    // Build cost matrix + log a; each thread owns rows n = tid*4 .. tid*4+3.
    for (int j = 0; j < 4; ++j) {
        const int n = (tid << 2) + j;
        const float* kr = kn + ((long long)b * 1024 + n) * 64;
        float dsum[8];
        for (int s = 0; s < 8; ++s) dsum[s] = 0.f;
        for (int c = 0; c < 64; ++c) {
            float kv = kr[c];
            for (int s = 0; s < 8; ++s) dsum[s] += kv * qs[s * 64 + c];
        }
        for (int s = 0; s < 8; ++s) Cs[n * 8 + s] = 1.f - dsum[s];
        loga[n] = logf(a[b * 1024 + n]);
        us[n] = 0.f;
    }
    __syncthreads();

    for (int it = 0; it < 5; ++it) {
        for (int si = 0; si < 5; ++si) {
            // ---- u[n] = log a - LSE_s(v - C) ----
            for (int j = 0; j < 4; ++j) {
                const int n = (tid << 2) + j;
                float t[8]; float mx = -1e30f;
                for (int s = 0; s < 8; ++s) { t[s] = vs[s] - Cs[n * 8 + s]; mx = fmaxf(mx, t[s]); }
                float se = 0.f;
                for (int s = 0; s < 8; ++s) se += expf(t[s] - mx);
                us[n] = loga[n] - (mx + logf(se));
            }
            __syncthreads();
            // ---- v[s] = log b - LSE_n(u - C) : wave shuffle + cross-wave ----
            float pm[8];
            for (int s = 0; s < 8; ++s) pm[s] = -1e30f;
            for (int j = 0; j < 4; ++j) {
                const int n = (tid << 2) + j;
                const float uu = us[n];
                for (int s = 0; s < 8; ++s) pm[s] = fmaxf(pm[s], uu - Cs[n * 8 + s]);
            }
            for (int s = 0; s < 8; ++s)
                for (int o = 16; o > 0; o >>= 1) pm[s] = fmaxf(pm[s], __shfl_xor(pm[s], o));
            if (lane == 0) for (int s = 0; s < 8; ++s) wred[wid][s] = pm[s];
            __syncthreads();
            if (tid < 8) {
                float m = wred[0][tid];
                for (int w = 1; w < 8; ++w) m = fmaxf(m, wred[w][tid]);
                msh[tid] = m;
            }
            __syncthreads();
            float ps[8];
            for (int s = 0; s < 8; ++s) ps[s] = 0.f;
            for (int j = 0; j < 4; ++j) {
                const int n = (tid << 2) + j;
                const float uu = us[n];
                for (int s = 0; s < 8; ++s) ps[s] += expf(uu - Cs[n * 8 + s] - msh[s]);
            }
            for (int s = 0; s < 8; ++s)
                for (int o = 16; o > 0; o >>= 1) ps[s] += __shfl_xor(ps[s], o);
            if (lane == 0) for (int s = 0; s < 8; ++s) wred[wid][s] = ps[s];
            __syncthreads();
            if (tid < 8) {
                float sm = 0.f;
                for (int w = 0; w < 8; ++w) sm += wred[w][tid];
                vs[tid] = logb[tid] - (msh[tid] + logf(sm));
            }
            __syncthreads();
        }
        if (it < 4) {
            // MESH entropy-gradient step (detached potentials):
            //   C -= pi*(log(pi+eps)+pi/(pi+eps))/B
            for (int j = 0; j < 4; ++j) {
                const int n = (tid << 2) + j;
                for (int s = 0; s < 8; ++s) {
                    const int i = n * 8 + s;
                    float p = expf(vs[s] + us[n] - Cs[i]);
                    Cs[i] -= p * (logf(p + 1e-8f) + p / (p + 1e-8f)) * (1.f / 16.f);
                }
            }
            __syncthreads();
        }
    }

    // Final transport plan: overwrite Cs with pi, emit attn[b,s,n].
    for (int j = 0; j < 4; ++j) {
        const int n = (tid << 2) + j;
        for (int s = 0; s < 8; ++s) {
            float p = expf(vs[s] + us[n] - Cs[n * 8 + s]);
            Cs[n * 8 + s] = p;
            attn[b * 8192 + s * 1024 + n] = p;
        }
    }
    __syncthreads();

    // upd[b,s,d] = sum_n pi[s,n] * vf[b,n,d]; each thread owns 2 (s,d) pairs.
    for (int pidx = tid; pidx < 512; pidx += 256) {
        const int s = pidx >> 6, d = pidx & 63;
        const float* vr = vf + (long long)b * 65536 + d;
        float acc = 0.f;
        for (int n = 0; n < 1024; ++n) acc += Cs[n * 8 + s] * vr[n * 64];
        upd[b * 512 + pidx] = acc;
    }
}

// ---------------------------------------------------------------------------
// Small elementwise / reduction kernels (f32 path)
// ---------------------------------------------------------------------------
__global__ void f32_to_f16_k(const float* __restrict__ x, _Float16* __restrict__ y, int n) {
    int i = blockIdx.x * blockDim.x + threadIdx.x;
    if (i < n) y[i] = (_Float16)x[i];
}
// Convert + zero-pad rows: src [M][K] f32 -> dst [M][Kp] f16.
__global__ void f32_to_f16_pad_k(const float* __restrict__ src, _Float16* __restrict__ dst,
                                 int M, int K, int Kp) {
    int i = blockIdx.x * blockDim.x + threadIdx.x;
    if (i >= M * Kp) return;
    int m = i / Kp, k = i - m * Kp;
    dst[i] = (_Float16)(k < K ? src[m * K + k] : 0.f);
}
__global__ void copy_f32_k(const float* __restrict__ x, float* __restrict__ y, int n) {
    int i = blockIdx.x * blockDim.x + threadIdx.x;
    if (i < n) y[i] = x[i];
}
// Snapshot slots: f32 copy + f16 convert in one pass.
__global__ void snapshot_slots_k(const float* __restrict__ x, float* __restrict__ yf,
                                 _Float16* __restrict__ yh, int n) {
    int i = blockIdx.x * blockDim.x + threadIdx.x;
    if (i < n) { float v = x[i]; yf[i] = v; yh[i] = (_Float16)v; }
}
__global__ void add_inplace_k(float* __restrict__ x, const float* __restrict__ y, int n) {
    int i = blockIdx.x * blockDim.x + threadIdx.x;
    if (i < n) x[i] += y[i];
}

// pos[y][x][c] = [y/31, x/31, 1-y/31, 1-x/31] . pos_w[c][:] + pos_b[c]
__global__ void build_pos_k(const float* __restrict__ pos_w, const float* __restrict__ pos_b,
                            float* __restrict__ pos) {
    int i = blockIdx.x * blockDim.x + threadIdx.x;
    if (i >= 32 * 32 * 64) return;
    int c = i & 63, x = (i >> 6) & 31, y = i >> 11;
    float gy = y * (1.f / 31.f), gx = x * (1.f / 31.f);
    pos[i] = gy * pos_w[c * 4 + 0] + gx * pos_w[c * 4 + 1] +
             (1.f - gy) * pos_w[c * 4 + 2] + (1.f - gx) * pos_w[c * 4 + 3] + pos_b[c];
}

// feat0[b][n][c] = x4[b][c][n] + pos[n][c]   (n = y*32+x)
__global__ void fuse_pos_k(const _Float16* __restrict__ x4, const float* __restrict__ pos,
                           _Float16* __restrict__ feat0) {
    int i = blockIdx.x * blockDim.x + threadIdx.x;
    if (i >= 16 * 1024 * 64) return;
    int c = i & 63, n = (i >> 6) & 1023, b = i >> 16;
    float v = (float)x4[(b * 64 + c) * 1024 + n] + pos[n * 64 + c];
    feat0[i] = (_Float16)v;
}

// LayerNorm over dim 64, one wave per row; writes f32 and optional f16.
__global__ void layernorm64_k(const float* __restrict__ x, const float* __restrict__ w,
                              const float* __restrict__ b, float* __restrict__ outF,
                              _Float16* __restrict__ outH, int R) {
    int row  = blockIdx.x * (blockDim.x >> 5) + (threadIdx.x >> 5);
    int lane = threadIdx.x & 31;
    if (row >= R) return;
    const float* xr = x + (long long)row * 64;
    float a0 = xr[lane], a1 = xr[lane + 32];
    float mean = warp_sum_all(a0 + a1) * (1.f / 64.f);
    float d0 = a0 - mean, d1 = a1 - mean;
    float var = warp_sum_all(d0 * d0 + d1 * d1) * (1.f / 64.f);
    float inv = rsqrtf(var + 1e-5f);
    float y0 = d0 * inv * w[lane] + b[lane];
    float y1 = d1 * inv * w[lane + 32] + b[lane + 32];
    if (outF) { outF[(long long)row * 64 + lane] = y0; outF[(long long)row * 64 + lane + 32] = y1; }
    if (outH) { outH[(long long)row * 64 + lane] = (_Float16)y0; outH[(long long)row * 64 + lane + 32] = (_Float16)y1; }
}

// L2-normalize rows of dim 64 (eps 1e-12), one wave per row.
__global__ void l2norm64_k(const float* __restrict__ x, float* __restrict__ y, int R) {
    int row  = blockIdx.x * (blockDim.x >> 5) + (threadIdx.x >> 5);
    int lane = threadIdx.x & 31;
    if (row >= R) return;
    const float* xr = x + (long long)row * 64;
    float a0 = xr[lane], a1 = xr[lane + 32];
    float nrm = sqrtf(warp_sum_all(a0 * a0 + a1 * a1));
    float inv = 1.f / fmaxf(nrm, 1e-12f);
    y[(long long)row * 64 + lane]      = a0 * inv;
    y[(long long)row * 64 + lane + 32] = a1 * inv;
}

// a[b,n] = softmax_n(inp[b,n,:].mwi_w + mwi_b) * 8 ; one block per b.
__global__ void softmax_a_k(const float* __restrict__ inp, const float* __restrict__ w,
                            const float* __restrict__ bb, float* __restrict__ a) {
    __shared__ float red[256];
    const int b = blockIdx.x, tid = threadIdx.x;
    float loc[4]; float mx = -1e30f;
    for (int j = 0; j < 4; ++j) {
        int n = tid * 4 + j;
        const float* r = inp + ((long long)b * 1024 + n) * 64;
        float s = bb[0];
        for (int c = 0; c < 64; ++c) s += r[c] * w[c];
        loc[j] = s; mx = fmaxf(mx, s);
    }
    red[tid] = mx; __syncthreads();
    for (int o = 128; o > 0; o >>= 1) { if (tid < o) red[tid] = fmaxf(red[tid], red[tid + o]); __syncthreads(); }
    mx = red[0]; __syncthreads();
    float se = 0.f;
    for (int j = 0; j < 4; ++j) se += expf(loc[j] - mx);
    red[tid] = se; __syncthreads();
    for (int o = 128; o > 0; o >>= 1) { if (tid < o) red[tid] += red[tid + o]; __syncthreads(); }
    float denom = red[0];
    for (int j = 0; j < 4; ++j)
        a[b * 1024 + tid * 4 + j] = expf(loc[j] - mx) / denom * 8.f;
}

// bm[b,s] = softmax_s(sln[b,s,:].mws_w + mws_b) * 8 ; one thread per b.
__global__ void bm_softmax_k(const float* __restrict__ sln, const float* __restrict__ w,
                             const float* __restrict__ bb, float* __restrict__ bm) {
    int b = threadIdx.x;
    if (b >= 16) return;
    float l[8]; float mx = -1e30f;
    for (int s = 0; s < 8; ++s) {
        const float* r = sln + (b * 8 + s) * 64;
        float t = bb[0];
        for (int c = 0; c < 64; ++c) t += r[c] * w[c];
        l[s] = t; mx = fmaxf(mx, t);
    }
    float se = 0.f;
    for (int s = 0; s < 8; ++s) se += expf(l[s] - mx);
    for (int s = 0; s < 8; ++s) bm[b * 8 + s] = expf(l[s] - mx) / se * 8.f;
}

__global__ void init_slots_k(const float* __restrict__ mu, const float* __restrict__ sg,
                             const float* __restrict__ noise, float* __restrict__ slots) {
    int i = blockIdx.x * blockDim.x + threadIdx.x;
    if (i >= 8192) return;
    int c = i & 63;
    slots[i] = mu[c] + (fabsf(sg[c]) + 1e-8f) * noise[i];
}

// GRU gate combine: slots = (1-z)*n + z*h
__global__ void gru_combine_k(const float* __restrict__ gi, const float* __restrict__ gh,
                              const float* __restrict__ hprev, float* __restrict__ slots) {
    int i = blockIdx.x * blockDim.x + threadIdx.x;
    if (i >= 8192) return;
    int row = i >> 6, c = i & 63;
    float ir = gi[row * 192 + c],        hr = gh[row * 192 + c];
    float iz = gi[row * 192 + 64 + c],   hz = gh[row * 192 + 64 + c];
    float in_ = gi[row * 192 + 128 + c], hn = gh[row * 192 + 128 + c];
    float r = 1.f / (1.f + expf(-(ir + hr)));
    float z = 1.f / (1.f + expf(-(iz + hz)));
    float nn = tanhf(in_ + r * hn);
    slots[i] = (1.f - z) * nn + z * hprev[i];
}

// ---------------------------------------------------------------------------
// Host orchestration
// ---------------------------------------------------------------------------
static inline int cdiv(long long a, long long b) { return (int)((a + b - 1) / b); }

extern "C" void kernel_launch(void* const* d_in, const int* in_sizes, int n_in,
                              void* d_out, int out_size, void* d_ws, size_t ws_size,
                              hipStream_t stream) {
    (void)in_sizes; (void)n_in; (void)out_size; (void)ws_size;
    const float* image    = (const float*)d_in[0];
    const float* noise    = (const float*)d_in[1];
    const float* conv1_w  = (const float*)d_in[2];  const float* conv1_b = (const float*)d_in[3];
    const float* conv2_w  = (const float*)d_in[4];  const float* conv2_b = (const float*)d_in[5];
    const float* conv3_w  = (const float*)d_in[6];  const float* conv3_b = (const float*)d_in[7];
    const float* conv4_w  = (const float*)d_in[8];  const float* conv4_b = (const float*)d_in[9];
    const float* pos_w    = (const float*)d_in[10]; const float* pos_b   = (const float*)d_in[11];
    const float* mlp1_w   = (const float*)d_in[12]; const float* mlp1_b  = (const float*)d_in[13];
    const float* mlp2_w   = (const float*)d_in[14]; const float* mlp2_b  = (const float*)d_in[15];
    const float* ln_in_w  = (const float*)d_in[16]; const float* ln_in_b = (const float*)d_in[17];
    const float* wk       = (const float*)d_in[18];
    const float* wv       = (const float*)d_in[19];
    const float* wq       = (const float*)d_in[20];
    const float* mwi_w    = (const float*)d_in[21]; const float* mwi_b   = (const float*)d_in[22];
    const float* mws_w    = (const float*)d_in[23]; const float* mws_b   = (const float*)d_in[24];
    const float* ln_sl_w  = (const float*)d_in[25]; const float* ln_sl_b = (const float*)d_in[26];
    const float* gru_wih  = (const float*)d_in[27]; const float* gru_whh = (const float*)d_in[28];
    const float* gru_bih  = (const float*)d_in[29]; const float* gru_bhh = (const float*)d_in[30];
    const float* ln_ff_w  = (const float*)d_in[31]; const float* ln_ff_b = (const float*)d_in[32];
    const float* fc1_w    = (const float*)d_in[33]; const float* fc1_b   = (const float*)d_in[34];
    const float* fc2_w    = (const float*)d_in[35]; const float* fc2_b   = (const float*)d_in[36];
    const float* slots_mu = (const float*)d_in[37]; const float* slots_sg = (const float*)d_in[38];

    // -------- workspace carve --------
    char* wsp = (char*)d_ws;
    size_t off = 0;
    auto alloc = [&](size_t elems, size_t esz) -> void* {
        void* p = wsp + off;
        off += (elems * esz + 255) & ~(size_t)255;
        return p;
    };
    _Float16* img_h   = (_Float16*)alloc(16 * 3 * 128 * 128, 2);
    _Float16* w1_h    = (_Float16*)alloc(64 * 96, 2);        // Kp(75) = 96
    _Float16* w2_h    = (_Float16*)alloc(64 * 1600, 2);      // 1600 % 32 == 0
    _Float16* w3_h    = (_Float16*)alloc(64 * 1600, 2);
    _Float16* w4_h    = (_Float16*)alloc(64 * 1600, 2);
    int4*     kt1     = (int4*)alloc(96, 16);
    int4*     kt2     = (int4*)alloc(1600, 16);
    int4*     kt3     = (int4*)alloc(1600, 16);
    int4*     kt4     = (int4*)alloc(1600, 16);
    _Float16* x1_h    = (_Float16*)alloc((size_t)16 * 64 * 128 * 128, 2);
    _Float16* x2_h    = (_Float16*)alloc((size_t)16 * 64 * 64 * 64, 2);
    _Float16* x3_h    = (_Float16*)alloc((size_t)16 * 64 * 32 * 32, 2);
    _Float16* x4_h    = (_Float16*)alloc((size_t)16 * 64 * 32 * 32, 2);
    float*    pos_f   = (float*)alloc(32 * 32 * 64, 4);
    _Float16* feat0_h = (_Float16*)alloc(16 * 1024 * 64, 2);
    _Float16* mlp1_h  = (_Float16*)alloc(64 * 64, 2);
    _Float16* mlp2_h  = (_Float16*)alloc(64 * 64, 2);
    _Float16* feat1_h = (_Float16*)alloc(16 * 1024 * 64, 2);
    float*    feat2_f = (float*)alloc(16 * 1024 * 64, 4);
    float*    inp_f   = (float*)alloc(16 * 1024 * 64, 4);
    _Float16* inp_h   = (_Float16*)alloc(16 * 1024 * 64, 2);
    _Float16* wk_h    = (_Float16*)alloc(64 * 64, 2);
    _Float16* wv_h    = (_Float16*)alloc(64 * 64, 2);
    _Float16* wq_h    = (_Float16*)alloc(64 * 64, 2);
    float*    k_f     = (float*)alloc(16 * 1024 * 64, 4);
    float*    v_f     = (float*)alloc(16 * 1024 * 64, 4);
    float*    kn_f    = (float*)alloc(16 * 1024 * 64, 4);
    float*    a_f     = (float*)alloc(16 * 1024, 4);
    float*    slots_f = (float*)alloc(8192, 4);
    float*    sprev_f = (float*)alloc(8192, 4);
    _Float16* sprev_h = (_Float16*)alloc(8192, 2);
    float*    sln_f   = (float*)alloc(8192, 4);
    _Float16* sln_h   = (_Float16*)alloc(8192, 2);
    float*    bm_f    = (float*)alloc(128, 4);
    float*    q_f     = (float*)alloc(8192, 4);
    float*    qn_f    = (float*)alloc(8192, 4);
    float*    upd_f   = (float*)alloc(8192, 4);
    _Float16* upd_h   = (_Float16*)alloc(8192, 2);
    _Float16* wih_h   = (_Float16*)alloc(192 * 64, 2);
    _Float16* whh_h   = (_Float16*)alloc(192 * 64, 2);
    float*    gi_f    = (float*)alloc(128 * 192, 4);
    float*    gh_f    = (float*)alloc(128 * 192, 4);
    float*    hln_f   = (float*)alloc(8192, 4);
    _Float16* hln_h   = (_Float16*)alloc(8192, 2);
    _Float16* fc1w_h  = (_Float16*)alloc(128 * 64, 2);
    _Float16* fc2w_h  = (_Float16*)alloc(64 * 128, 2);
    _Float16* fc1o_h  = (_Float16*)alloc(128 * 128, 2);
    float*    fc2o_f  = (float*)alloc(128 * 64, 4);

    float* slots_out = (float*)d_out;          // [16,8,64]
    float* attn_out  = (float*)d_out + 8192;   // [16,8,1024]

    // -------- weight & input conversions + conv gather tables --------
    auto cvt = [&](const float* src, _Float16* dst, int n) {
        f32_to_f16_k<<<cdiv(n, 256), 256, 0, stream>>>(src, dst, n);
    };
    cvt(image, img_h, 16 * 3 * 128 * 128);
    f32_to_f16_pad_k<<<cdiv(64 * 96, 256), 256, 0, stream>>>(conv1_w, w1_h, 64, 75, 96);
    cvt(conv2_w, w2_h, 64 * 1600);
    cvt(conv3_w, w3_h, 64 * 1600);
    cvt(conv4_w, w4_h, 64 * 1600);
    cvt(mlp1_w, mlp1_h, 64 * 64);
    cvt(mlp2_w, mlp2_h, 64 * 64);
    cvt(wk, wk_h, 64 * 64);
    cvt(wv, wv_h, 64 * 64);
    cvt(wq, wq_h, 64 * 64);
    cvt(gru_wih, wih_h, 192 * 64);
    cvt(gru_whh, whh_h, 192 * 64);
    cvt(fc1_w, fc1w_h, 128 * 64);
    cvt(fc2_w, fc2w_h, 64 * 128);
    build_ktab_k<<<1, 96, 0, stream>>>(128, 128, 75, 96, kt1);
    build_ktab_k<<<cdiv(1600, 256), 256, 0, stream>>>(128, 128, 1600, 1600, kt2);
    build_ktab_k<<<cdiv(1600, 256), 256, 0, stream>>>(64, 64, 1600, 1600, kt3);
    build_ktab_k<<<cdiv(1600, 256), 256, 0, stream>>>(32, 32, 1600, 1600, kt4);

    // -------- CNN encoder (WMMA implicit GEMM, 64xN strips) --------
    conv5x5_wmma<<<cdiv(16384, 8), 256, 0, stream>>>(img_h, w1_h, conv1_b, kt1, x1_h,
                                                      16, 3, 128, 128, 1, 128, 128, 96);
    conv5x5_wmma<<<cdiv(4096, 8), 256, 0, stream>>>(x1_h, w2_h, conv2_b, kt2, x2_h,
                                                     16, 64, 128, 128, 2, 64, 64, 1600);
    conv5x5_wmma<<<cdiv(1024, 8), 256, 0, stream>>>(x2_h, w3_h, conv3_b, kt3, x3_h,
                                                     16, 64, 64, 64, 2, 32, 32, 1600);
    conv5x5_wmma<<<cdiv(1024, 8), 256, 0, stream>>>(x3_h, w4_h, conv4_b, kt4, x4_h,
                                                     16, 64, 32, 32, 1, 32, 32, 1600);

    // -------- position embed + feature MLP --------
    build_pos_k<<<cdiv(32 * 32 * 64, 256), 256, 0, stream>>>(pos_w, pos_b, pos_f);
    fuse_pos_k<<<cdiv(16 * 1024 * 64, 256), 256, 0, stream>>>(x4_h, pos_f, feat0_h);
    gemm_wmma<<<128, 256, 0, stream>>>(feat0_h, mlp1_h, mlp1_b, nullptr, feat1_h,
                                        16384, 64, 64, 1);
    gemm_wmma<<<128, 256, 0, stream>>>(feat1_h, mlp2_h, mlp2_b, feat2_f, nullptr,
                                        16384, 64, 64, 0);
    layernorm64_k<<<cdiv(16384 * 32, 256), 256, 0, stream>>>(feat2_f, ln_in_w, ln_in_b,
                                                              inp_f, inp_h, 16384);
    gemm_wmma<<<128, 256, 0, stream>>>(inp_h, wk_h, nullptr, k_f, nullptr, 16384, 64, 64, 0);
    gemm_wmma<<<128, 256, 0, stream>>>(inp_h, wv_h, nullptr, v_f, nullptr, 16384, 64, 64, 0);
    l2norm64_k<<<cdiv(16384 * 32, 256), 256, 0, stream>>>(k_f, kn_f, 16384);
    softmax_a_k<<<16, 256, 0, stream>>>(inp_f, mwi_w, mwi_b, a_f);
    init_slots_k<<<32, 256, 0, stream>>>(slots_mu, slots_sg, noise, slots_f);

    // -------- slot attention iterations --------
    for (int it = 0; it < 3; ++it) {
        snapshot_slots_k<<<32, 256, 0, stream>>>(slots_f, sprev_f, sprev_h, 8192);
        layernorm64_k<<<16, 256, 0, stream>>>(slots_f, ln_sl_w, ln_sl_b, sln_f, sln_h, 128);
        bm_softmax_k<<<1, 16, 0, stream>>>(sln_f, mws_w, mws_b, bm_f);
        gemm_wmma<<<1, 256, 0, stream>>>(sln_h, wq_h, nullptr, q_f, nullptr, 128, 64, 64, 0);
        l2norm64_k<<<16, 256, 0, stream>>>(q_f, qn_f, 128);
        // Fused: cost + MESH(4 grad steps) + 25 warm-started Sinkhorn iters +
        // final pi -> attn, and upd = attn . v   (one block per batch)
        mesh_sinkhorn_k<<<16, 256, 0, stream>>>(kn_f, qn_f, a_f, bm_f, v_f, attn_out, upd_f);
        f32_to_f16_k<<<32, 256, 0, stream>>>(upd_f, upd_h, 8192);
        gemm_wmma<<<3, 256, 0, stream>>>(upd_h, wih_h, gru_bih, gi_f, nullptr, 128, 64, 192, 0);
        gemm_wmma<<<3, 256, 0, stream>>>(sprev_h, whh_h, gru_bhh, gh_f, nullptr, 128, 64, 192, 0);
        gru_combine_k<<<32, 256, 0, stream>>>(gi_f, gh_f, sprev_f, slots_f);
        layernorm64_k<<<16, 256, 0, stream>>>(slots_f, ln_ff_w, ln_ff_b, hln_f, hln_h, 128);
        gemm_wmma<<<2, 256, 0, stream>>>(hln_h, fc1w_h, fc1_b, nullptr, fc1o_h, 128, 64, 128, 1);
        gemm_wmma<<<1, 256, 0, stream>>>(fc1o_h, fc2w_h, fc2_b, fc2o_f, nullptr, 128, 128, 64, 0);
        add_inplace_k<<<32, 256, 0, stream>>>(slots_f, fc2o_f, 8192);
    }
    copy_f32_k<<<32, 256, 0, stream>>>(slots_f, slots_out, 8192);
}
